// CALText_14250701488551
// MI455X (gfx1250) — compile-verified, matrix operand order checked by reading, not run on hardware
//
#include <hip/hip_runtime.h>
#include <hip/hip_bf16.h>

typedef __attribute__((ext_vector_type(16))) _Float16 v16h;
typedef __attribute__((ext_vector_type(8)))  _Float16 v8h;
typedef __attribute__((ext_vector_type(8)))  float    v8f;

// Problem constants
#define Tn   64
#define Bn   32
#define Hh   16
#define Ww   64
#define HID  256
#define WDIM 256
#define CDIM 684
#define TDIM 128
#define ADIM 512
#define COVD 128
#define BHW  (Bn*Hh*Ww)      // 32768
#define HWn  (Hh*Ww)         // 1024

// Branch-free tanh: one v_exp_f32 + one v_rcp_f32, no EXEC divergence.
__device__ __forceinline__ float tanh_fast(float x) {
    float ax = fabsf(x);
    float e  = __expf(-2.0f * ax);
    float r  = (1.0f - e) * __builtin_amdgcn_rcpf(1.0f + e);
    return copysignf(r, x);
}

__device__ __forceinline__ float sigmoid_fast(float x) {
    return __builtin_amdgcn_rcpf(1.0f + __expf(-x));
}

// ---------------- init ----------------
__global__ void k_init(float* ap) {
    int i = blockIdx.x*256 + threadIdx.x;
    if (i < BHW) ap[i] = 0.0f;
}

// ---------------- anno_mean ----------------
__global__ void k_annomean(const float* __restrict__ anno, const float* __restrict__ a_m,
                           float* __restrict__ am) {
    int idx = blockIdx.x*256 + threadIdx.x;
    if (idx >= Bn*CDIM) return;
    int b = idx / CDIM, c = idx % CDIM;
    float s = 0.f, ms = 0.f;
    for (int i = 0; i < HWn; ++i) {
        float m = a_m[b*HWn + i];
        s  += anno[((size_t)b*HWn + i)*CDIM + c] * m;
        ms += m;
    }
    am[idx] = s / ms;
}

// ---------------- h0 = tanh(am @ Wa2h + ba2h) ----------------
__global__ void k_h0(const float* __restrict__ am, const float* __restrict__ Wa2h,
                     const float* __restrict__ ba2h, float* __restrict__ hcur) {
    int idx = blockIdx.x*256 + threadIdx.x;
    if (idx >= Bn*HID) return;
    int b = idx >> 8, i = idx & 255;
    float s = ba2h[i];
    for (int c = 0; c < CDIM; ++c) s += am[b*CDIM + c] * Wa2h[c*HID + i];
    hcur[idx] = tanhf(s);
}

// ---------------- emb_shift gather ----------------
__global__ void k_embshift(const int* __restrict__ cost_y, const float* __restrict__ embed,
                           float* __restrict__ emb_shift) {
    int idx = blockIdx.x*256 + threadIdx.x;
    if (idx >= Tn*Bn*WDIM) return;
    int t = idx >> 13;            // Bn*WDIM = 8192
    int rem = idx & 8191;
    int b = rem >> 8, w = rem & 255;
    float v = 0.f;
    if (t > 0) { int y = cost_y[(t-1)*Bn + b]; v = embed[y*WDIM + w]; }
    emb_shift[idx] = v;
}

// ---------------- ctx_att = anno @ Ua (one-time, plain) ----------------
__global__ __launch_bounds__(256) void k_ctxatt(const float* __restrict__ anno,
                                                const float* __restrict__ Ua,
                                                float* __restrict__ ctx_att) {
    int row = blockIdx.x;                 // 0..BHW-1
    int j = threadIdx.x;                  // 0..255
    __shared__ float ar[CDIM];
    for (int c = j; c < CDIM; c += 256) ar[c] = anno[(size_t)row*CDIM + c];
    __syncthreads();
    float s0 = 0.f, s1 = 0.f;
    for (int c = 0; c < CDIM; ++c) {
        float a = ar[c];
        s0 += a * Ua[c*ADIM + j];
        s1 += a * Ua[c*ADIM + j + 256];
    }
    ctx_att[(size_t)row*ADIM + j]       = s0;
    ctx_att[(size_t)row*ADIM + j + 256] = s1;
}

// ---------------- swizzle Uf -> f16 B-fragment layout ----------------
// ufw[((nt*4+ks)*32+lane)*16 + h] = Uf[ ks*32 + Kof(lane,h) ][ nt*16 + lane%16 ]
__global__ void k_ufw(const float* __restrict__ Uf, _Float16* __restrict__ ufw) {
    int idx = blockIdx.x*256 + threadIdx.x;     // 32*4*32*16 = 65536
    if (idx >= 65536) return;
    int h    = idx & 15;
    int lane = (idx >> 4) & 31;
    int ksnt = idx >> 9;
    int ks = ksnt & 3, nt = ksnt >> 2;
    int K = ks*32 + ((h < 8) ? h : h + 8) + ((lane >= 16) ? 8 : 0);
    int n = nt*16 + (lane & 15);
    ufw[idx] = (_Float16)Uf[K*ADIM + n];
}

// ---------------- GRU1: pre_h and hWa = pre_h @ Wa ----------------
__global__ __launch_bounds__(256) void k_gru1(
    const float* __restrict__ hcur, const float* __restrict__ emb_t,
    const float* __restrict__ W_yz_yr, const float* __restrict__ b_yz_yr,
    const float* __restrict__ U_hz_hr, const float* __restrict__ W_yh,
    const float* __restrict__ b_yh, const float* __restrict__ U_rh,
    const float* __restrict__ Wa, float* __restrict__ preh, float* __restrict__ hWa) {
    int b = blockIdx.x, j = threadIdx.x;
    __shared__ float h[HID], em[WDIM], ph[HID];
    h[j]  = hcur[b*HID + j];
    em[j] = emb_t[b*WDIM + j];
    __syncthreads();
    float z0 = b_yz_yr[j], z1v = b_yz_yr[j + 256];
    float t2 = 0.f, t3 = b_yh[j];
    for (int k = 0; k < HID; ++k) {
        float ek = em[k], hk = h[k];
        z0  += ek * W_yz_yr[k*512 + j]       + hk * U_hz_hr[k*512 + j];
        z1v += ek * W_yz_yr[k*512 + j + 256] + hk * U_hz_hr[k*512 + j + 256];
        t2  += hk * U_rh[k*HID + j];
        t3  += ek * W_yh[k*HID + j];
    }
    float r1 = sigmoid_fast(z0);
    float z1 = sigmoid_fast(z1v);
    float p = z1*h[j] + (1.f - z1)*tanh_fast(r1*t2 + t3);
    ph[j] = p;
    preh[b*HID + j] = p;
    __syncthreads();
    float a0 = 0.f, a1 = 0.f;
    for (int k = 0; k < HID; ++k) {
        float pk = ph[k];
        a0 += pk * Wa[k*ADIM + j];
        a1 += pk * Wa[k*ADIM + j + 256];
    }
    hWa[b*ADIM + j]       = a0;
    hWa[b*ADIM + j + 256] = a1;
}

// ---------------- coverage conv 5x5 SAME -> f16 cover ----------------
__global__ __launch_bounds__(128) void k_cover(const float* __restrict__ ap,
                                               const float* __restrict__ conv_Q,
                                               _Float16* __restrict__ cover16) {
    int bhw = blockIdx.x;            // 0..BHW-1
    int d = threadIdx.x;             // 0..127
    int b = bhw >> 10;
    int hw = bhw & 1023;
    int hh = hw >> 6, ww = hw & 63;
    __shared__ float patch[25];
    if (d < 25) {
        int kh = d / 5, kw = d % 5;
        int y = hh + kh - 2, x = ww + kw - 2;
        patch[d] = (y >= 0 && y < Hh && x >= 0 && x < Ww) ? ap[(b<<10) + (y<<6) + x] : 0.f;
    }
    __syncthreads();
    float s = 0.f;
#pragma unroll
    for (int k = 0; k < 25; ++k) s += patch[k] * conv_Q[k*COVD + d];
    cover16[(size_t)bhw*COVD + d] = (_Float16)s;
}

// ---------------- WMMA energy kernel ----------------
// E = tanh(ctx_att + hWa + cover@Uf) . Va   (reduce over ADIM=512)
// block = 128 threads (4 waves); 16-row M-tile; wave handles 8 N-tiles of 16.
__global__ __launch_bounds__(128) void k_energy(
    const _Float16* __restrict__ cover16, const _Float16* __restrict__ ufw,
    const float* __restrict__ ctx_att, const float* __restrict__ hWa,
    const float* __restrict__ Va, float* __restrict__ e_out) {
    __shared__ float rows[16];
    int tid  = threadIdx.x;
    int lane = tid & 31;
    int wave = tid >> 5;
    int m0 = blockIdx.x * 16;
    int bb = blockIdx.x >> 6;               // 64 tiles per batch row block
    int mrow = m0 + (lane & 15);
    int kbase = (lane >= 16) ? 8 : 0;
    int mhalf = (lane >= 16) ? 8 : 0;

    // Load 4 A-fragments (K = 0..127 in steps of 32), ISA 16-bit A layout.
    v16h a[4];
    const _Float16* arow = cover16 + (size_t)mrow * COVD;
#pragma unroll
    for (int ks = 0; ks < 4; ++ks) {
        const v8h* p0 = (const v8h*)(arow + ks*32 + kbase);
        const v8h* p1 = (const v8h*)(arow + ks*32 + kbase + 16);
        v8h lo = *p0, hi = *p1;
#pragma unroll
        for (int h = 0; h < 8; ++h) { a[ks][h] = lo[h]; a[ks][h+8] = hi[h]; }
    }
    if (tid < 16) rows[tid] = 0.0f;
    __syncthreads();

    float pr[8];
#pragma unroll
    for (int r = 0; r < 8; ++r) pr[r] = 0.0f;

#pragma unroll
    for (int t = 0; t < 8; ++t) {
        int nt = wave * 8 + t;
        v8f acc = {};
#pragma unroll
        for (int ks = 0; ks < 4; ++ks) {
            const v8h* q0 = (const v8h*)(ufw + ((size_t)(nt*4 + ks)*32 + lane)*16);
            v8h blo = q0[0], bhi = q0[1];
            v16h bf;
#pragma unroll
            for (int h = 0; h < 8; ++h) { bf[h] = blo[h]; bf[h+8] = bhi[h]; }
            acc = __builtin_amdgcn_wmma_f32_16x16x32_f16(
                false, a[ks], false, bf, (short)0, acc, false, false);
        }
        int n = nt*16 + (lane & 15);
        float van = Va[n];
        float hw  = hWa[bb*ADIM + n];
#pragma unroll
        for (int r = 0; r < 8; ++r) {
            int m = mhalf + r;
            float v = acc[r] + ctx_att[(size_t)(m0 + m)*ADIM + n] + hw;
            pr[r] += tanh_fast(v) * van;
        }
    }
    // Butterfly-reduce each row partial across the 16-lane half-group
    // (lane-permute hardware, no LDS traffic), then 2 lanes/wave hit LDS.
#pragma unroll
    for (int r = 0; r < 8; ++r) {
        float v = pr[r];
#pragma unroll
        for (int off = 8; off >= 1; off >>= 1) v += __shfl_xor(v, off, 16);
        if ((lane & 15) == 0) atomicAdd(&rows[mhalf + r], v);
    }
    __syncthreads();
    if (tid < 16) e_out[m0 + tid] = rows[tid];
}

// ---------------- softmax over (H,W) per batch; alpha; ap += alpha ----------------
__global__ __launch_bounds__(256) void k_softmax(const float* __restrict__ e,
                                                 const float* __restrict__ a_m,
                                                 float* __restrict__ alpha,
                                                 float* __restrict__ ap) {
    int b = blockIdx.x, tid = threadIdx.x;
    __shared__ float red[256];
    float mx = -3.4e38f;
    for (int i = tid; i < HWn; i += 256) mx = fmaxf(mx, e[b*HWn + i]);
    red[tid] = mx; __syncthreads();
    for (int s = 128; s > 0; s >>= 1) { if (tid < s) red[tid] = fmaxf(red[tid], red[tid+s]); __syncthreads(); }
    mx = red[0]; __syncthreads();
    float sm = 0.f;
    for (int i = tid; i < HWn; i += 256) sm += __expf(e[b*HWn + i] - mx) * a_m[b*HWn + i];
    red[tid] = sm; __syncthreads();
    for (int s = 128; s > 0; s >>= 1) { if (tid < s) red[tid] += red[tid+s]; __syncthreads(); }
    sm = red[0];
    float rs = __builtin_amdgcn_rcpf(sm);
    for (int i = tid; i < HWn; i += 256) {
        float al = __expf(e[b*HWn + i] - mx) * a_m[b*HWn + i] * rs;
        alpha[b*HWn + i] = al;
        ap[b*HWn + i] += al;
    }
}

// ---------------- context = alpha . anno ----------------
__global__ __launch_bounds__(256) void k_context(const float* __restrict__ alpha,
                                                 const float* __restrict__ anno,
                                                 float* __restrict__ ctxbuf,
                                                 float* __restrict__ call_t) {
    int b = blockIdx.x;
    int c = blockIdx.y*256 + threadIdx.x;
    if (c >= CDIM) return;
    float s = 0.f;
    const float* ab = alpha + b*HWn;
    const float* an = anno + (size_t)b*HWn*CDIM + c;
    for (int i = 0; i < HWn; ++i) s += ab[i] * an[(size_t)i*CDIM];
    ctxbuf[b*CDIM + c] = s;
    call_t[b*CDIM + c] = s;
}

// ---------------- GRU2: h update with y_m mask ----------------
__global__ __launch_bounds__(256) void k_gru2(
    const float* __restrict__ preh, const float* __restrict__ ctxbuf,
    const float* __restrict__ ym_t,
    const float* __restrict__ U_hz_hr_nl, const float* __restrict__ b_hz_hr_nl,
    const float* __restrict__ W_c_z_r, const float* __restrict__ U_rh_nl,
    const float* __restrict__ b_rh_nl, const float* __restrict__ W_c_h_nl,
    float* __restrict__ hcur, float* __restrict__ hall_t) {
    int b = blockIdx.x, j = threadIdx.x;
    __shared__ float ph[HID];
    __shared__ float cx[CDIM];
    ph[j] = preh[b*HID + j];
    for (int c = j; c < CDIM; c += 256) cx[c] = ctxbuf[b*CDIM + c];
    __syncthreads();
    float z0 = b_hz_hr_nl[j], z1v = b_hz_hr_nl[j + 256];
    float t2 = b_rh_nl[j], t3 = 0.f;
    for (int k = 0; k < HID; ++k) {
        float pk = ph[k];
        z0  += pk * U_hz_hr_nl[k*512 + j];
        z1v += pk * U_hz_hr_nl[k*512 + j + 256];
        t2  += pk * U_rh_nl[k*HID + j];
    }
    for (int c = 0; c < CDIM; ++c) {
        float cc = cx[c];
        z0  += cc * W_c_z_r[c*512 + j];
        z1v += cc * W_c_z_r[c*512 + j + 256];
        t3  += cc * W_c_h_nl[c*HID + j];
    }
    float r2 = sigmoid_fast(z0);
    float z2 = sigmoid_fast(z1v);
    float ht = z2*ph[j] + (1.f - z2)*tanh_fast(r2*t2 + t3);
    float ym = ym_t[b];
    float hprev = hcur[b*HID + j];
    ht = ym*ht + (1.f - ym)*hprev;
    hcur[b*HID + j]   = ht;
    hall_t[b*HID + j] = ht;
}

// ---------------- epilogue: fused logits + pair-maxpool ----------------
__global__ __launch_bounds__(128) void k_pooled(
    const float* __restrict__ hall, const float* __restrict__ call,
    const float* __restrict__ embshift,
    const float* __restrict__ Wh, const float* __restrict__ bh,
    const float* __restrict__ Wc, const float* __restrict__ bc,
    const float* __restrict__ Wy, const float* __restrict__ by,
    float* __restrict__ pooled) {
    int tb = blockIdx.x;
    int p = threadIdx.x;
    __shared__ float hrow[HID], erow[WDIM], crow[CDIM];
    for (int i = p; i < HID; i += 128) {
        hrow[i] = hall[(size_t)tb*HID + i];
        erow[i] = embshift[(size_t)tb*WDIM + i];
    }
    for (int c = p; c < CDIM; c += 128) crow[c] = call[(size_t)tb*CDIM + c];
    __syncthreads();
    float acc[2];
#pragma unroll
    for (int q = 0; q < 2; ++q) {
        int j = 2*p + q;
        float s = bh[j] + bc[j] + by[j];
        for (int i = 0; i < HID; ++i) s += hrow[i]*Wh[i*WDIM + j] + erow[i]*Wy[i*WDIM + j];
        for (int c = 0; c < CDIM; ++c) s += crow[c]*Wc[c*WDIM + j];
        acc[q] = s;
    }
    pooled[tb*128 + p] = fmaxf(acc[0], acc[1]);
}

__global__ __launch_bounds__(128) void k_logit2(const float* __restrict__ pooled,
                                                const float* __restrict__ Wo,
                                                const float* __restrict__ bo,
                                                float* __restrict__ logit2) {
    int tb = blockIdx.x, o = threadIdx.x;
    __shared__ float pr[128];
    pr[o] = pooled[tb*128 + o];
    __syncthreads();
    float s = bo[o];
    for (int p = 0; p < 128; ++p) s += pr[p]*Wo[p*TDIM + o];
    logit2[tb*TDIM + o] = s;
}

__global__ __launch_bounds__(128) void k_nll(const float* __restrict__ logit2,
                                             const int* __restrict__ cost_y,
                                             const float* __restrict__ y_m,
                                             float* __restrict__ nllbuf) {
    int tb = blockIdx.x, tid = threadIdx.x;
    __shared__ float red[128];
    float v = logit2[tb*TDIM + tid];
    red[tid] = v; __syncthreads();
    for (int s = 64; s > 0; s >>= 1) { if (tid < s) red[tid] = fmaxf(red[tid], red[tid+s]); __syncthreads(); }
    float mx = red[0]; __syncthreads();
    red[tid] = expf(v - mx); __syncthreads();
    for (int s = 64; s > 0; s >>= 1) { if (tid < s) red[tid] += red[tid+s]; __syncthreads(); }
    if (tid == 0) {
        int y = cost_y[tb];
        float lse = mx + logf(red[0]);
        float lp = logit2[tb*TDIM + y] - lse;
        nllbuf[tb] = -lp * y_m[tb];
    }
}

__global__ __launch_bounds__(256) void k_finalreduce(const float* __restrict__ nllbuf,
                                                     float* __restrict__ out) {
    __shared__ float red[256];
    int tid = threadIdx.x;
    float s = 0.f;
    for (int i = tid; i < Tn*Bn; i += 256) s += nllbuf[i];
    red[tid] = s; __syncthreads();
    for (int st = 128; st > 0; st >>= 1) { if (tid < st) red[tid] += red[tid+st]; __syncthreads(); }
    if (tid == 0) out[0] = red[0] / (float)Bn;
}

// ---------------- orchestration ----------------
extern "C" void kernel_launch(void* const* d_in, const int* in_sizes, int n_in,
                              void* d_out, int out_size, void* d_ws, size_t ws_size,
                              hipStream_t stream) {
    const float* anno       = (const float*)d_in[0];
    const float* a_m        = (const float*)d_in[1];
    const float* y_m        = (const float*)d_in[2];
    const float* embed      = (const float*)d_in[3];
    const float* Wa2h       = (const float*)d_in[4];
    const float* ba2h       = (const float*)d_in[5];
    const float* Wc         = (const float*)d_in[6];
    const float* bc         = (const float*)d_in[7];
    const float* Wh         = (const float*)d_in[8];
    const float* bh         = (const float*)d_in[9];
    const float* Wy         = (const float*)d_in[10];
    const float* by         = (const float*)d_in[11];
    const float* Wo         = (const float*)d_in[12];
    const float* bo         = (const float*)d_in[13];
    const float* W_yz_yr    = (const float*)d_in[14];
    const float* b_yz_yr    = (const float*)d_in[15];
    const float* U_hz_hr    = (const float*)d_in[16];
    const float* W_yh       = (const float*)d_in[17];
    const float* b_yh       = (const float*)d_in[18];
    const float* U_rh       = (const float*)d_in[19];
    const float* U_hz_hr_nl = (const float*)d_in[20];
    const float* b_hz_hr_nl = (const float*)d_in[21];
    const float* W_c_z_r    = (const float*)d_in[22];
    const float* U_rh_nl    = (const float*)d_in[23];
    const float* b_rh_nl    = (const float*)d_in[24];
    const float* W_c_h_nl   = (const float*)d_in[25];
    const float* Wa         = (const float*)d_in[26];
    const float* Ua         = (const float*)d_in[27];
    const float* Va         = (const float*)d_in[28];
    const float* conv_Q     = (const float*)d_in[29];
    const float* Uf         = (const float*)d_in[30];
    const int*   cost_y     = (const int*)d_in[31];

    float* wsf = (float*)d_ws;
    // workspace layout (float units, 256B aligned blocks)
    size_t O_CTX   = 0;                                 // 32768*512
    size_t O_COVER = O_CTX   + (size_t)BHW*ADIM;        // 32768*128 f16 -> /2
    size_t O_UFW   = O_COVER + (size_t)BHW*COVD/2;      // 65536 f16 -> /2
    size_t O_EMB   = O_UFW   + 65536/2;
    size_t O_AM    = O_EMB   + (size_t)Tn*Bn*WDIM;
    size_t O_HCUR  = O_AM    + 21952;
    size_t O_PREH  = O_HCUR  + Bn*HID;
    size_t O_HWA   = O_PREH  + Bn*HID;
    size_t O_E     = O_HWA   + Bn*ADIM;
    size_t O_ALPHA = O_E     + BHW;
    size_t O_AP    = O_ALPHA + BHW;
    size_t O_CTXB  = O_AP    + BHW;
    size_t O_HALL  = O_CTXB  + 21952;
    size_t O_CALL  = O_HALL  + (size_t)Tn*Bn*HID;
    size_t O_POOL  = O_CALL  + (size_t)Tn*Bn*CDIM;
    size_t O_LG2   = O_POOL  + (size_t)Tn*Bn*128;
    size_t O_NLL   = O_LG2   + (size_t)Tn*Bn*TDIM;

    _Float16* cover16 = (_Float16*)(wsf + O_COVER);
    _Float16* ufw     = (_Float16*)(wsf + O_UFW);

    // ---- one-time precompute ----
    k_init<<<(BHW+255)/256, 256, 0, stream>>>(wsf + O_AP);
    k_annomean<<<(Bn*CDIM+255)/256, 256, 0, stream>>>(anno, a_m, wsf + O_AM);
    k_h0<<<(Bn*HID+255)/256, 256, 0, stream>>>(wsf + O_AM, Wa2h, ba2h, wsf + O_HCUR);
    k_embshift<<<(Tn*Bn*WDIM+255)/256, 256, 0, stream>>>(cost_y, embed, wsf + O_EMB);
    k_ctxatt<<<BHW, 256, 0, stream>>>(anno, Ua, wsf + O_CTX);
    k_ufw<<<65536/256, 256, 0, stream>>>(Uf, ufw);

    // ---- sequential scan ----
    for (int t = 0; t < Tn; ++t) {
        const float* emb_t = wsf + O_EMB + (size_t)t*Bn*WDIM;
        k_gru1<<<Bn, 256, 0, stream>>>(wsf + O_HCUR, emb_t, W_yz_yr, b_yz_yr,
                                       U_hz_hr, W_yh, b_yh, U_rh, Wa,
                                       wsf + O_PREH, wsf + O_HWA);
        k_cover<<<BHW, 128, 0, stream>>>(wsf + O_AP, conv_Q, cover16);
        k_energy<<<BHW/16, 128, 0, stream>>>(cover16, ufw, wsf + O_CTX,
                                             wsf + O_HWA, Va, wsf + O_E);
        k_softmax<<<Bn, 256, 0, stream>>>(wsf + O_E, a_m, wsf + O_ALPHA, wsf + O_AP);
        dim3 cg(Bn, (CDIM + 255)/256);
        k_context<<<cg, 256, 0, stream>>>(wsf + O_ALPHA, anno, wsf + O_CTXB,
                                          wsf + O_CALL + (size_t)t*Bn*CDIM);
        k_gru2<<<Bn, 256, 0, stream>>>(wsf + O_PREH, wsf + O_CTXB, y_m + t*Bn,
                                       U_hz_hr_nl, b_hz_hr_nl, W_c_z_r,
                                       U_rh_nl, b_rh_nl, W_c_h_nl,
                                       wsf + O_HCUR, wsf + O_HALL + (size_t)t*Bn*HID);
    }

    // ---- epilogue ----
    k_pooled<<<Tn*Bn, 128, 0, stream>>>(wsf + O_HALL, wsf + O_CALL, wsf + O_EMB,
                                        Wh, bh, Wc, bc, Wy, by, wsf + O_POOL);
    k_logit2<<<Tn*Bn, 128, 0, stream>>>(wsf + O_POOL, Wo, bo, wsf + O_LG2);
    k_nll<<<Tn*Bn, 128, 0, stream>>>(wsf + O_LG2, cost_y, y_m, wsf + O_NLL);
    k_finalreduce<<<1, 256, 0, stream>>>(wsf + O_NLL, (float*)d_out);
}
